// GraphT_670014898257
// MI455X (gfx1250) — matrix operand order repaired and checked by reading, not compiled, not run on hardware
//
#include <hip/hip_runtime.h>
#include <hip/hip_bf16.h>
#include <math.h>

// ---------------------------------------------------------------------------
// Types for CDNA5 WMMA
// ---------------------------------------------------------------------------
typedef __attribute__((ext_vector_type(16))) __bf16 v16bf;
typedef __attribute__((ext_vector_type(8)))  float  v8f;
typedef __bf16 bf16x2 __attribute__((ext_vector_type(2)));

union FragU { unsigned u[8]; uint4 q[2]; v16bf v; };

#define GD 256
#define ATT_SCALE 0.0625f   // GD^-0.5

// packed fp32x2 -> bf16x2 via hardware cvt (RNE)
__device__ __forceinline__ unsigned pack_bf16(float x, float y) {
  bf16x2 p;
  p.x = (__bf16)x;
  p.y = (__bf16)y;
  return __builtin_bit_cast(unsigned, p);
}

// ---------------------------------------------------------------------------
// Weight pre-conversion: fp32 -> bf16 (row-major preserved)
// ---------------------------------------------------------------------------
__global__ void __launch_bounds__(256)
cvt_bf16_kernel(const float* __restrict__ in, unsigned short* __restrict__ out,
                int n) {
  int i = (blockIdx.x * blockDim.x + threadIdx.x) * 2;
  if (i >= n) return;
  float2 v = *reinterpret_cast<const float2*>(in + i);
  *reinterpret_cast<unsigned*>(out + i) = pack_bf16(v.x, v.y);
}

// ---------------------------------------------------------------------------
// Generic GEMM: C[M,Nc] = epilogue(A[M,K] * W[Nc,K]^T + bias), W pre-cast bf16
//   mode 0: + bias
//   mode 1: gelu(x + bias)   (exact erf)
//   mode 2: x + bias + res1 + res2
// Block: 256 threads (8 waves). Block tile: 32 rows x 256 cols.
// Each wave: 2 row-tiles x 2 col-tiles = 4 WMMA per K-step (B frags reused).
// LDS double buffered: one barrier per K-step; staging rows clamped to M-1
// (out-of-range rows compute garbage that is never stored).
// ---------------------------------------------------------------------------
__global__ void __launch_bounds__(256)
gemm_bf16_wmma(const float* __restrict__ A, const unsigned short* __restrict__ W,
               const float* __restrict__ bias, float* __restrict__ C,
               int M, int K, int Nc, int mode,
               const float* __restrict__ res1, const float* __restrict__ res2) {
  __shared__ unsigned ldsA[1024];          // 2 x (32 rows x 16 dwords)

  const int lane = threadIdx.x & 31;
  const int wave = threadIdx.x >> 5;
  const int row0 = blockIdx.x * 32;
  const int cb   = blockIdx.y * 256;
  const int r    = lane & 15;
  const int hs   = lane >> 4;

  // bf16 weight row pointers for the two col-tiles owned by this wave
  const unsigned short* wr0 = W + (size_t)(cb + wave * 16 + r) * K + hs * 16;
  const unsigned short* wr1 = W + (size_t)(cb + 128 + wave * 16 + r) * K + hs * 16;

  // A staging: thread t stages dwords t (rows 0-15) and t+256 (rows 16-31);
  // row index clamped so no branch is needed.
  const int sr = threadIdx.x >> 4;         // 0..15
  const int sd = threadIdx.x & 15;         // K-pair index
  const int ra0 = min(row0 + sr, M - 1);
  const int ra1 = min(row0 + 16 + sr, M - 1);
  const float* ap0 = A + (size_t)ra0 * K + 2 * sd;
  const float* ap1 = A + (size_t)ra1 * K + 2 * sd;

  // prologue: stage kb = 0 into buffer 0
  {
    float2 a0 = *reinterpret_cast<const float2*>(ap0);
    float2 a1 = *reinterpret_cast<const float2*>(ap1);
    ldsA[threadIdx.x]       = pack_bf16(a0.x, a0.y);
    ldsA[threadIdx.x + 256] = pack_bf16(a1.x, a1.y);
  }

  v8f acc00 = {};  // rows 0-15,  cols tile0
  v8f acc01 = {};  // rows 0-15,  cols tile1
  v8f acc10 = {};  // rows 16-31, cols tile0
  v8f acc11 = {};  // rows 16-31, cols tile1

  int p = 0;
  for (int kb = 0; kb < K; kb += 32, p ^= 1) {
    __syncthreads();   // orders stage(i) -> read(i+1) and read(i) -> stage(i+1)

    const unsigned* buf = ldsA + p * 512;

    // A fragments: two aligned b128 LDS reads each
    // frag dword g holds K = hs*16 + 2g
    FragU fa0, fa1;
    {
      const uint4* l0 = reinterpret_cast<const uint4*>(buf + r * 16);
      const uint4* l1 = reinterpret_cast<const uint4*>(buf + (16 + r) * 16);
      fa0.q[0] = l0[2 * hs];
      fa0.q[1] = l0[2 * hs + 1];
      fa1.q[0] = l1[2 * hs];
      fa1.q[1] = l1[2 * hs + 1];
    }

    // B fragments: contiguous 16 bf16 per lane -> two global b128 reads each
    // frag dword g holds K = kb + hs*16 + 2g
    FragU fb0, fb1;
    {
      const uint4* p0 = reinterpret_cast<const uint4*>(wr0 + kb);
      const uint4* p1 = reinterpret_cast<const uint4*>(wr1 + kb);
      fb0.q[0] = p0[0];
      fb0.q[1] = p0[1];
      fb1.q[0] = p1[0];
      fb1.q[1] = p1[1];
    }

    // stage next K-slice into the other buffer (overlaps with WMMA)
    if (kb + 32 < K) {
      float2 a0 = *reinterpret_cast<const float2*>(ap0 + kb + 32);
      float2 a1 = *reinterpret_cast<const float2*>(ap1 + kb + 32);
      unsigned* nb = ldsA + (p ^ 1) * 512;
      nb[threadIdx.x]       = pack_bf16(a0.x, a0.y);
      nb[threadIdx.x + 256] = pack_bf16(a1.x, a1.y);
    }

    acc00 = __builtin_amdgcn_wmma_f32_16x16x32_bf16(false, fa0.v, false, fb0.v,
                                                    (short)0, acc00, false, false);
    acc01 = __builtin_amdgcn_wmma_f32_16x16x32_bf16(false, fa0.v, false, fb1.v,
                                                    (short)0, acc01, false, false);
    acc10 = __builtin_amdgcn_wmma_f32_16x16x32_bf16(false, fa1.v, false, fb0.v,
                                                    (short)0, acc10, false, false);
    acc11 = __builtin_amdgcn_wmma_f32_16x16x32_bf16(false, fa1.v, false, fb1.v,
                                                    (short)0, acc11, false, false);
  }

  // C/D layout: VGPR i -> M = i (lanes 0-15) or 8+i (lanes 16-31); N = lane&15
  const int mb = hs * 8;
#pragma unroll
  for (int rt = 0; rt < 2; ++rt) {
#pragma unroll
    for (int ct = 0; ct < 2; ++ct) {
      const v8f& acc = rt == 0 ? (ct == 0 ? acc00 : acc01)
                               : (ct == 0 ? acc10 : acc11);
      const int col = cb + ct * 128 + wave * 16 + r;
#pragma unroll
      for (int i = 0; i < 8; ++i) {
        int row = row0 + rt * 16 + mb + i;
        if (row >= M) continue;
        float c = acc[i] + (bias ? bias[col] : 0.f);
        if (mode == 1) {
          c = 0.5f * c * (1.f + erff(c * 0.70710678118654752f));
        } else if (mode == 2) {
          size_t o = (size_t)row * Nc + col;
          c += res1[o] + res2[o];
        }
        C[(size_t)row * Nc + col] = c;
      }
    }
  }
}

// ---------------------------------------------------------------------------
// LayerNorm over rows of 256: one wave32 per row
// ---------------------------------------------------------------------------
__global__ void __launch_bounds__(256)
ln_kernel(const float* __restrict__ in, const float* __restrict__ w,
          const float* __restrict__ b, float* __restrict__ out, int n) {
  const int gwave = (blockIdx.x * blockDim.x + threadIdx.x) >> 5;
  const int lane  = threadIdx.x & 31;
  if (gwave >= n) return;
  const float* row = in + (size_t)gwave * GD;
  float4 a = *reinterpret_cast<const float4*>(row + lane * 8);
  float4 c = *reinterpret_cast<const float4*>(row + lane * 8 + 4);
  float sum = a.x + a.y + a.z + a.w + c.x + c.y + c.z + c.w;
  float sq  = a.x*a.x + a.y*a.y + a.z*a.z + a.w*a.w
            + c.x*c.x + c.y*c.y + c.z*c.z + c.w*c.w;
#pragma unroll
  for (int m = 16; m >= 1; m >>= 1) {
    sum += __shfl_xor(sum, m, 32);
    sq  += __shfl_xor(sq,  m, 32);
  }
  float mu  = sum * (1.f / 256.f);
  float var = sq * (1.f / 256.f) - mu * mu;
  float rs  = rsqrtf(var + 1e-5f);
  float* orow = out + (size_t)gwave * GD;
  float vals[8] = {a.x, a.y, a.z, a.w, c.x, c.y, c.z, c.w};
#pragma unroll
  for (int j = 0; j < 8; ++j) {
    int cidx = lane * 8 + j;
    orow[cidx] = (vals[j] - mu) * rs * w[cidx] + b[cidx];
  }
}

// ---------------------------------------------------------------------------
// Edge softmax + aggregation (atomics)
// ---------------------------------------------------------------------------
__device__ __forceinline__ unsigned ordkey(float f) {
  unsigned u = __float_as_uint(f);
  return (u & 0x80000000u) ? ~u : (u | 0x80000000u);
}
__device__ __forceinline__ float ordval(unsigned k) {
  unsigned u = (k & 0x80000000u) ? (k ^ 0x80000000u) : ~k;
  return __uint_as_float(u);
}

__global__ void __launch_bounds__(256)
init_kernel(unsigned* __restrict__ amax, float* __restrict__ z,
            float* __restrict__ rst, int n) {
  int i = blockIdx.x * blockDim.x + threadIdx.x;
  if (i >= n) return;
  amax[i] = 0u;           // encodes -inf under ordkey ordering
  z[i] = 0.f;
  rst[i] = 0.f;
}

__global__ void __launch_bounds__(256)
edge_max(const int* __restrict__ src, const int* __restrict__ dst,
         const float* __restrict__ qkv, unsigned* __restrict__ amax, int E) {
  int idx = blockIdx.x * blockDim.x + threadIdx.x;
  if (idx >= E * 64) return;
  int e = idx >> 6;
  int c = (idx & 63) << 2;
  int s = src[e], d = dst[e];
  float4 q = *reinterpret_cast<const float4*>(qkv + (size_t)s * 768 + c);
  float4 k = *reinterpret_cast<const float4*>(qkv + (size_t)d * 768 + 256 + c);
  unsigned* base = amax + (size_t)d * GD + c;
  atomicMax(base + 0, ordkey(q.x * k.x * ATT_SCALE));
  atomicMax(base + 1, ordkey(q.y * k.y * ATT_SCALE));
  atomicMax(base + 2, ordkey(q.z * k.z * ATT_SCALE));
  atomicMax(base + 3, ordkey(q.w * k.w * ATT_SCALE));
}

__global__ void __launch_bounds__(256)
edge_sum(const int* __restrict__ src, const int* __restrict__ dst,
         const float* __restrict__ qkv, const unsigned* __restrict__ amax,
         float* __restrict__ z, int E) {
  int idx = blockIdx.x * blockDim.x + threadIdx.x;
  if (idx >= E * 64) return;
  int e = idx >> 6;
  int c = (idx & 63) << 2;
  int s = src[e], d = dst[e];
  float4 q = *reinterpret_cast<const float4*>(qkv + (size_t)s * 768 + c);
  float4 k = *reinterpret_cast<const float4*>(qkv + (size_t)d * 768 + 256 + c);
  const unsigned* mb = amax + (size_t)d * GD + c;
  float* zb = z + (size_t)d * GD + c;
  atomicAdd(zb + 0, __expf(q.x * k.x * ATT_SCALE - ordval(mb[0])));
  atomicAdd(zb + 1, __expf(q.y * k.y * ATT_SCALE - ordval(mb[1])));
  atomicAdd(zb + 2, __expf(q.z * k.z * ATT_SCALE - ordval(mb[2])));
  atomicAdd(zb + 3, __expf(q.w * k.w * ATT_SCALE - ordval(mb[3])));
}

__global__ void __launch_bounds__(256)
edge_agg(const int* __restrict__ src, const int* __restrict__ dst,
         const float* __restrict__ qkv, const unsigned* __restrict__ amax,
         const float* __restrict__ z, float* __restrict__ rst, int E) {
  int idx = blockIdx.x * blockDim.x + threadIdx.x;
  if (idx >= E * 64) return;
  int e = idx >> 6;
  int c = (idx & 63) << 2;
  int s = src[e], d = dst[e];
  float4 q = *reinterpret_cast<const float4*>(qkv + (size_t)s * 768 + c);
  float4 k = *reinterpret_cast<const float4*>(qkv + (size_t)d * 768 + 256 + c);
  float4 v = *reinterpret_cast<const float4*>(qkv + (size_t)s * 768 + 512 + c);
  const unsigned* mb = amax + (size_t)d * GD + c;
  const float* zb = z + (size_t)d * GD + c;
  float* rb = rst + (size_t)d * GD + c;
  atomicAdd(rb + 0, v.x * __expf(q.x * k.x * ATT_SCALE - ordval(mb[0])) / zb[0]);
  atomicAdd(rb + 1, v.y * __expf(q.y * k.y * ATT_SCALE - ordval(mb[1])) / zb[1]);
  atomicAdd(rb + 2, v.z * __expf(q.z * k.z * ATT_SCALE - ordval(mb[2])) / zb[2]);
  atomicAdd(rb + 3, v.w * __expf(q.w * k.w * ATT_SCALE - ordval(mb[3])) / zb[3]);
}

// ---------------------------------------------------------------------------
// Launch
// ---------------------------------------------------------------------------
extern "C" void kernel_launch(void* const* d_in, const int* in_sizes, int n_in,
                              void* d_out, int out_size, void* d_ws, size_t ws_size,
                              hipStream_t stream) {
  const float* x      = (const float*)d_in[0];
  const float* h      = (const float*)d_in[1];
  const int*   src    = (const int*)d_in[2];
  const int*   dst    = (const int*)d_in[3];
  const float* conv_w = (const float*)d_in[4];
  const float* conv_b = (const float*)d_in[5];
  const float* norm_w = (const float*)d_in[6];
  const float* norm_b = (const float*)d_in[7];
  const float* nin_w  = (const float*)d_in[8];
  const float* nin_b  = (const float*)d_in[9];
  const float* w_qkv  = (const float*)d_in[10];
  const float* w_out  = (const float*)d_in[11];
  const float* b_out  = (const float*)d_in[12];
  const float* ffn_w  = (const float*)d_in[13];
  const float* ffn_b  = (const float*)d_in[14];
  const float* w1     = (const float*)d_in[15];
  const float* b1     = (const float*)d_in[16];
  const float* w2     = (const float*)d_in[17];
  const float* b2     = (const float*)d_in[18];

  const int N = in_sizes[1] / GD;
  const int E = in_sizes[2];
  const size_t NC = (size_t)N * GD;

  float* ws      = (float*)d_ws;
  float* xp      = ws + 0 * NC;
  float* hn      = ws + 1 * NC;
  float* s       = ws + 2 * NC;
  float* qkv     = ws + 3 * NC;              // N x 768
  unsigned* amax = (unsigned*)(ws + 6 * NC);
  float* z       = ws + 7 * NC;
  float* rst     = ws + 8 * NC;
  float* attn    = ws + 9 * NC;
  float* fln     = ws + 10 * NC;
  float* hbuf    = ws + 11 * NC;             // N x 512

  // bf16 weight scratch after hbuf
  unsigned short* wbf     = (unsigned short*)(ws + 13 * NC);
  unsigned short* cw_bf   = wbf;
  unsigned short* wqkv_bf = cw_bf   + in_sizes[4];
  unsigned short* wout_bf = wqkv_bf + in_sizes[10];
  unsigned short* w1_bf   = wout_bf + in_sizes[11];
  unsigned short* w2_bf   = w1_bf   + in_sizes[15];

  const dim3 blk(256);
  const int mtiles   = (N + 31) / 32;
  const int lnBlocks = (N * 32 + 255) / 256;
  const int eBlocks  = (E * 64 + 255) / 256;
  const int iBlocks  = ((int)NC + 255) / 256;

  // pre-convert weights to bf16
  cvt_bf16_kernel<<<(in_sizes[4]/2 + 255)/256,  blk, 0, stream>>>(conv_w, cw_bf,   in_sizes[4]);
  cvt_bf16_kernel<<<(in_sizes[10]/2 + 255)/256, blk, 0, stream>>>(w_qkv,  wqkv_bf, in_sizes[10]);
  cvt_bf16_kernel<<<(in_sizes[11]/2 + 255)/256, blk, 0, stream>>>(w_out,  wout_bf, in_sizes[11]);
  cvt_bf16_kernel<<<(in_sizes[15]/2 + 255)/256, blk, 0, stream>>>(w1,     w1_bf,   in_sizes[15]);
  cvt_bf16_kernel<<<(in_sizes[17]/2 + 255)/256, blk, 0, stream>>>(w2,     w2_bf,   in_sizes[17]);

  // hn = LN(h); s = LN(hn)
  ln_kernel<<<lnBlocks, blk, 0, stream>>>(h,  norm_w, norm_b, hn, N);
  ln_kernel<<<lnBlocks, blk, 0, stream>>>(hn, nin_w,  nin_b,  s,  N);

  // xp = x @ conv_w^T + conv_b   (K=4096, Nc=256)
  gemm_bf16_wmma<<<dim3(mtiles, 1), blk, 0, stream>>>(
      x, cw_bf, conv_b, xp, N, 4096, 256, 0, nullptr, nullptr);

  // qkv = s @ w_qkv^T            (K=256, Nc=768, no bias)
  gemm_bf16_wmma<<<dim3(mtiles, 3), blk, 0, stream>>>(
      s, wqkv_bf, nullptr, qkv, N, 256, 768, 0, nullptr, nullptr);

  // edge softmax + aggregation
  init_kernel<<<iBlocks, blk, 0, stream>>>(amax, z, rst, (int)NC);
  edge_max<<<eBlocks, blk, 0, stream>>>(src, dst, qkv, amax, E);
  edge_sum<<<eBlocks, blk, 0, stream>>>(src, dst, qkv, amax, z, E);
  edge_agg<<<eBlocks, blk, 0, stream>>>(src, dst, qkv, amax, z, rst, E);

  // attn = rst @ w_out^T + b_out (K=256, Nc=256)
  gemm_bf16_wmma<<<dim3(mtiles, 1), blk, 0, stream>>>(
      rst, wout_bf, b_out, attn, N, 256, 256, 0, nullptr, nullptr);

  // fln = LN(attn)
  ln_kernel<<<lnBlocks, blk, 0, stream>>>(attn, ffn_w, ffn_b, fln, N);

  // hbuf = gelu(fln @ w1^T + b1) (K=256, Nc=512)
  gemm_bf16_wmma<<<dim3(mtiles, 2), blk, 0, stream>>>(
      fln, w1_bf, b1, hbuf, N, 256, 512, 1, nullptr, nullptr);

  // out = hbuf @ w2^T + b2 + hn + xp  (K=512, Nc=256)
  gemm_bf16_wmma<<<dim3(mtiles, 1), blk, 0, stream>>>(
      hbuf, w2_bf, b2, (float*)d_out, N, 512, 256, 2, hn, xp);
}
